// GASConv_29678224016150
// MI455X (gfx1250) — matrix operand order repaired and checked by reading, not compiled, not run on hardware
//
#include <hip/hip_runtime.h>
#include <hip/hip_bf16.h>

typedef __bf16 bf16;
typedef __attribute__((ext_vector_type(16))) __bf16 v16bf;
typedef __attribute__((ext_vector_type(8)))  __bf16 v8bf;
typedef __attribute__((ext_vector_type(8)))  float  v8f;

// ---------------------------------------------------------------------------
// Utility kernels
// ---------------------------------------------------------------------------
__global__ void k_fill(float* p, size_t n, float v) {
    size_t i  = (size_t)blockIdx.x * blockDim.x + threadIdx.x;
    size_t st = (size_t)gridDim.x * blockDim.x;
    for (; i < n; i += st) p[i] = v;
}

// Pack f32 row-major weight W[Kdim][Ndim] into WMMA bf16 B-fragment order:
// chunk ((n/16)*(Kdim/32) + k/32) holds 32 lanes x 16 contiguous bf16.
// B-matrix layout (16-bit, 32x16): lane = (n%16) + (((k%32)/8)%2)*16, elem = k%8 + ((k%32)/16)*8.
__global__ void k_pack_w(const float* __restrict__ W, bf16* __restrict__ Wp,
                         int Kdim, int Ndim) {
    int tid = blockIdx.x * blockDim.x + threadIdx.x;
    if (tid >= Kdim * Ndim) return;
    int k = tid / Ndim, n = tid % Ndim;
    int kr   = k & 31;
    int lane = (n & 15) + (((kr >> 3) & 1) << 4);
    int e    = (kr & 7) + ((kr >> 4) << 3);
    size_t idx = ((size_t)((n >> 4) * (Kdim >> 5) + (k >> 5))) * 512 + lane * 16 + e;
    Wp[idx] = (bf16)W[(size_t)k * Ndim + n];
}

__device__ __forceinline__ void atomicMaxF(float* addr, float val) {
    if (val >= 0.0f) atomicMax((int*)addr, __float_as_int(val));
    else             atomicMin((unsigned int*)addr, __float_as_uint(val));
}

// Build 16x32 bf16 A fragment for one lane from an LDS tile (row stride LDK).
// A layout: lane l holds row m=l%16; elems 0..7 -> K = kb + (l/16)*8 .. +7,
// elems 8..15 -> K = kb + 16 + (l/16)*8 .. +7.
template<int LDK>
__device__ __forceinline__ v16bf load_a_frag(const bf16* sA, int mrow, int hf, int kb) {
    v8bf c0 = *(const v8bf*)(sA + mrow * LDK + kb + hf * 8);
    v8bf c1 = *(const v8bf*)(sA + mrow * LDK + kb + 16 + hf * 8);
    v16bf a;
#pragma unroll
    for (int i = 0; i < 8; ++i) { a[i] = c0[i]; a[i + 8] = c1[i]; }
    return a;
}

// ---------------------------------------------------------------------------
// Generic node GEMM: out[row, col_off + n] = relu?( X[16xKDIM] @ Wp + bias )
// block = 128 threads (4 waves); grid = (Nnodes/16, ntiles/4)
// ---------------------------------------------------------------------------
template<int KDIM>
__global__ void k_gemm_node(const float* __restrict__ X, const bf16* __restrict__ Wp,
                            const float* __restrict__ bias, float* __restrict__ out,
                            int ldo, int col_off, int do_relu) {
    constexpr int LDK = KDIM + 8;               // pad: 4-bank shift per row
    __shared__ __attribute__((aligned(32))) bf16 sA[16 * LDK];
    const int tid  = threadIdx.x;
    const int row0 = blockIdx.x * 16;
    for (int i = tid; i < 16 * KDIM; i += 128) {
        int m = i / KDIM, k = i % KDIM;
        sA[m * LDK + k] = (bf16)X[(size_t)(row0 + m) * KDIM + k];
    }
    __syncthreads();

    const int wave = tid >> 5, lane = tid & 31;
    const int hf = lane >> 4, l15 = lane & 15;
    const int ntile = blockIdx.y * 4 + wave;
    constexpr int KS = KDIM / 32;

    v8f acc = {};
    const bf16* wp = Wp + (size_t)ntile * KS * 512 + lane * 16;
#pragma unroll
    for (int ks = 0; ks < KS; ++ks) {
        v16bf a = load_a_frag<LDK>(sA, l15, hf, ks * 32);
        v16bf b = *(const v16bf*)(wp + ks * 512);
        acc = __builtin_amdgcn_wmma_f32_16x16x32_bf16(false, a, false, b,
                                                      (short)0, acc, false, false);
    }
    const int n  = ntile * 16 + l15;            // C/D: N = lane%16
    const float bv = bias[n];
#pragma unroll
    for (int r = 0; r < 8; ++r) {
        int m = r + hf * 8;                     // C/D: M = r + 8*(lane/16)
        float v = acc[r] + bv;
        if (do_relu) v = fmaxf(v, 0.0f);
        out[(size_t)(row0 + m) * ldo + col_off + n] = v;
    }
}

// ---------------------------------------------------------------------------
// he = relu(e_feat @ We + be + uW[src] + vW[dst])
// block = 256 threads (8 waves), 16 edges x 256 cols per block
// ---------------------------------------------------------------------------
__global__ void k_edge_he(const float* __restrict__ e_feat, const int* __restrict__ src,
                          const int* __restrict__ dst, const bf16* __restrict__ WeP,
                          const float* __restrict__ be, const float* __restrict__ uW,
                          const float* __restrict__ vW, float* __restrict__ he) {
    constexpr int LDK = 136;
    __shared__ __attribute__((aligned(32))) bf16 sA[16 * LDK];
    __shared__ int sS[16], sD[16];
    const int tid = threadIdx.x;
    const size_t e0 = (size_t)blockIdx.x * 16;
    for (int i = tid; i < 16 * 128; i += 256) {
        int m = i >> 7, k = i & 127;
        sA[m * LDK + k] = (bf16)e_feat[(e0 + m) * 128 + k];
    }
    if (tid < 16) { sS[tid] = src[e0 + tid]; sD[tid] = dst[e0 + tid]; }
    __syncthreads();

    const int wave = tid >> 5, lane = tid & 31;
    const int hf = lane >> 4, l15 = lane & 15;
    v8f acc0 = {}, acc1 = {};
    const bf16* wp0 = WeP + (size_t)(2 * wave + 0) * 4 * 512 + lane * 16;
    const bf16* wp1 = WeP + (size_t)(2 * wave + 1) * 4 * 512 + lane * 16;
#pragma unroll
    for (int ks = 0; ks < 4; ++ks) {
        v16bf a  = load_a_frag<LDK>(sA, l15, hf, ks * 32);
        v16bf b0 = *(const v16bf*)(wp0 + ks * 512);
        v16bf b1 = *(const v16bf*)(wp1 + ks * 512);
        acc0 = __builtin_amdgcn_wmma_f32_16x16x32_bf16(false, a, false, b0,
                                                       (short)0, acc0, false, false);
        acc1 = __builtin_amdgcn_wmma_f32_16x16x32_bf16(false, a, false, b1,
                                                       (short)0, acc1, false, false);
    }
    const int n0 = (2 * wave + 0) * 16 + l15;
    const int n1 = (2 * wave + 1) * 16 + l15;
#pragma unroll
    for (int r = 0; r < 8; ++r) {
        int m = r + hf * 8;
        size_t row = (e0 + m) * 256;
        size_t su = (size_t)sS[m] * 256, dv = (size_t)sD[m] * 256;
        float x0 = acc0[r] + be[n0] + uW[su + n0] + vW[dv + n0];
        float x1 = acc1[r] + be[n1] + uW[su + n1] + vW[dv + n1];
        he[row + n0] = fmaxf(x0, 0.0f);
        he[row + n1] = fmaxf(x1, 0.0f);
    }
}

// ---------------------------------------------------------------------------
// Per-edge attention dots + segment max. One wave per edge (8 edges/block).
// ---------------------------------------------------------------------------
__global__ void k_edge_dots(const float* __restrict__ e_feat, const float* __restrict__ u_feat,
                            const float* __restrict__ v_feat, const int* __restrict__ src,
                            const int* __restrict__ dst, const float* __restrict__ att_u,
                            const float* __restrict__ att_v, float* __restrict__ edb,
                            float* __restrict__ edf, float* __restrict__ mb,
                            float* __restrict__ mf) {
    const size_t edge = (size_t)blockIdx.x * 8 + (threadIdx.x >> 5);
    const int lane = threadIdx.x & 31;
    const int s = src[edge], d = dst[edge];
    const float* au = att_u + (size_t)s * 256;
    const float* av = att_v + (size_t)d * 256;
    const float* vf = v_feat + (size_t)d * 128;
    const float* uf = u_feat + (size_t)s * 128;
    const float* ef = e_feat + edge * 128;
    float ab = 0.0f, af = 0.0f;
#pragma unroll
    for (int j = 0; j < 4; ++j) {
        int c = lane + 32 * j;
        float ev = ef[c];
        ab = fmaf(vf[c], au[c], ab);
        ab = fmaf(ev, au[128 + c], ab);
        af = fmaf(uf[c], av[c], af);
        af = fmaf(ev, av[128 + c], af);
    }
#pragma unroll
    for (int off = 16; off; off >>= 1) {
        ab += __shfl_xor(ab, off);
        af += __shfl_xor(af, off);
    }
    if (lane == 0) {
        edb[edge] = ab; edf[edge] = af;
        atomicMaxF(mb + s, ab);
        atomicMaxF(mf + d, af);
    }
}

// exp(logit - segmax) and segment sums
__global__ void k_exp_sum(const int* __restrict__ src, const int* __restrict__ dst,
                          float* __restrict__ edb, float* __restrict__ edf,
                          const float* __restrict__ mb, const float* __restrict__ mf,
                          float* __restrict__ sb, float* __restrict__ sf, int NE) {
    int e = blockIdx.x * blockDim.x + threadIdx.x;
    if (e >= NE) return;
    int s = src[e], d = dst[e];
    float wb = __expf(edb[e] - mb[s]);
    float wf = __expf(edf[e] - mf[d]);
    edb[e] = wb; edf[e] = wf;
    atomicAdd(sb + s, wb);
    atomicAdd(sf + d, wf);
}

// weighted scatter-sum aggregation. One wave per edge.
__global__ void k_aggregate(const float* __restrict__ e_feat, const float* __restrict__ u_feat,
                            const float* __restrict__ v_feat, const int* __restrict__ src,
                            const int* __restrict__ dst, const float* __restrict__ edb,
                            const float* __restrict__ edf, const float* __restrict__ sb,
                            const float* __restrict__ sf, float* __restrict__ agg_u,
                            float* __restrict__ agg_v) {
    const size_t edge = (size_t)blockIdx.x * 8 + (threadIdx.x >> 5);
    const int lane = threadIdx.x & 31;
    const int s = src[edge], d = dst[edge];
    const float wb = edb[edge] / sb[s];
    const float wf = edf[edge] / sf[d];
    const float* vf = v_feat + (size_t)d * 128;
    const float* uf = u_feat + (size_t)s * 128;
    const float* ef = e_feat + edge * 128;
    float* au = agg_u + (size_t)s * 256;
    float* av = agg_v + (size_t)d * 256;
#pragma unroll
    for (int j = 0; j < 4; ++j) {
        int c = lane + 32 * j;
        float ev = ef[c];
        atomicAdd(au + c,       vf[c] * wb);
        atomicAdd(au + 128 + c, ev * wb);
        atomicAdd(av + c,       uf[c] * wf);
        atomicAdd(av + 128 + c, ev * wf);
    }
}

// ---------------------------------------------------------------------------
extern "C" void kernel_launch(void* const* d_in, const int* in_sizes, int n_in,
                              void* d_out, int out_size, void* d_ws, size_t ws_size,
                              hipStream_t stream) {
    (void)n_in; (void)out_size; (void)ws_size;
    const float* e_feat = (const float*)d_in[0];
    const float* u_feat = (const float*)d_in[1];
    const float* v_feat = (const float*)d_in[2];
    const int*   src    = (const int*)d_in[3];
    const int*   dst    = (const int*)d_in[4];
    const float* We  = (const float*)d_in[5];  const float* be  = (const float*)d_in[6];
    const float* Wu  = (const float*)d_in[7];  const float* bu  = (const float*)d_in[8];
    const float* Wv  = (const float*)d_in[9];  const float* bv  = (const float*)d_in[10];
    const float* Wau = (const float*)d_in[11]; const float* bau = (const float*)d_in[12];
    const float* Wav = (const float*)d_in[13]; const float* bav = (const float*)d_in[14];
    const float* Wnu = (const float*)d_in[15]; const float* bnu = (const float*)d_in[16];
    const float* Wnv = (const float*)d_in[17]; const float* bnv = (const float*)d_in[18];
    const float* Vu  = (const float*)d_in[19]; const float* bVu = (const float*)d_in[20];
    const float* Vv  = (const float*)d_in[21]; const float* bVv = (const float*)d_in[22];

    const int NE = in_sizes[3];
    const int NU = in_sizes[1] / 128;
    const int NV = in_sizes[2] / 128;

    // workspace carve-up (f32 region, then bf16 packed weights; 32B aligned)
    float* f = (float*)d_ws;
    float* uW    = f; f += (size_t)NU * 256;
    float* vW    = f; f += (size_t)NV * 256;
    float* att_u = f; f += (size_t)NU * 256;
    float* att_v = f; f += (size_t)NV * 256;
    float* edb   = f; f += NE;
    float* edf   = f; f += NE;
    float* agg_u = f; f += (size_t)NU * 256;   // zero region starts here
    float* agg_v = f; f += (size_t)NV * 256;
    float* sb    = f; f += NU;
    float* sf    = f; f += NV;                 // ...ends here
    float* mb    = f; f += NU;                 // -inf region
    float* mf    = f; f += NV;
    bf16* bp   = (bf16*)f;
    bf16* WeP  = bp; bp += 128 * 256;
    bf16* WuP  = bp; bp += 128 * 256;
    bf16* WvP  = bp; bp += 128 * 256;
    bf16* WauP = bp; bp += 128 * 256;
    bf16* WavP = bp; bp += 128 * 256;
    bf16* WnuP = bp; bp += 256 * 128;
    bf16* WnvP = bp; bp += 256 * 128;
    bf16* VuP  = bp; bp += 128 * 128;
    bf16* VvP  = bp; bp += 128 * 128;

    float* he = (float*)d_out;
    float* hu = he + (size_t)NE * 256;
    float* hv = hu + (size_t)NU * 256;

    // 1) pack weights to bf16 WMMA fragment order
    auto pg = [](int K, int N) { return (K * N + 255) / 256; };
    k_pack_w<<<pg(128,256), 256, 0, stream>>>(We,  WeP,  128, 256);
    k_pack_w<<<pg(128,256), 256, 0, stream>>>(Wu,  WuP,  128, 256);
    k_pack_w<<<pg(128,256), 256, 0, stream>>>(Wv,  WvP,  128, 256);
    k_pack_w<<<pg(128,256), 256, 0, stream>>>(Wau, WauP, 128, 256);
    k_pack_w<<<pg(128,256), 256, 0, stream>>>(Wav, WavP, 128, 256);
    k_pack_w<<<pg(256,128), 256, 0, stream>>>(Wnu, WnuP, 256, 128);
    k_pack_w<<<pg(256,128), 256, 0, stream>>>(Wnv, WnvP, 256, 128);
    k_pack_w<<<pg(128,128), 256, 0, stream>>>(Vu,  VuP,  128, 128);
    k_pack_w<<<pg(128,128), 256, 0, stream>>>(Vv,  VvP,  128, 128);

    // 2) init accumulators / softmax state
    k_fill<<<2048, 256, 0, stream>>>(agg_u, (size_t)(NU + NV) * 256 + NU + NV, 0.0f);
    k_fill<<<64, 256, 0, stream>>>(mb, (size_t)(NU + NV), -__builtin_huge_valf());

    // 3) node linears (WMMA bf16, K=128)
    k_gemm_node<128><<<dim3(NU/16, 4), 128, 0, stream>>>(u_feat, WuP,  bu,  uW,    256, 0,   0);
    k_gemm_node<128><<<dim3(NV/16, 4), 128, 0, stream>>>(v_feat, WvP,  bv,  vW,    256, 0,   0);
    k_gemm_node<128><<<dim3(NU/16, 4), 128, 0, stream>>>(u_feat, WauP, bau, att_u, 256, 0,   0);
    k_gemm_node<128><<<dim3(NV/16, 4), 128, 0, stream>>>(v_feat, WavP, bav, att_v, 256, 0,   0);
    k_gemm_node<128><<<dim3(NU/16, 2), 128, 0, stream>>>(u_feat, VuP,  bVu, hu,    256, 0,   0);
    k_gemm_node<128><<<dim3(NV/16, 2), 128, 0, stream>>>(v_feat, VvP,  bVv, hv,    256, 0,   0);

    // 4) edge GEMM + gather + relu -> he
    k_edge_he<<<NE/16, 256, 0, stream>>>(e_feat, src, dst, WeP, be, uW, vW, he);

    // 5) attention logits + segment softmax + aggregation
    k_edge_dots<<<NE/8, 256, 0, stream>>>(e_feat, u_feat, v_feat, src, dst,
                                          att_u, att_v, edb, edf, mb, mf);
    k_exp_sum<<<(NE + 255) / 256, 256, 0, stream>>>(src, dst, edb, edf, mb, mf, sb, sf, NE);
    k_aggregate<<<NE/8, 256, 0, stream>>>(e_feat, u_feat, v_feat, src, dst,
                                          edb, edf, sb, sf, agg_u, agg_v);

    // 6) aggregated node GEMMs (K=256) with relu -> second halves of hu/hv
    k_gemm_node<256><<<dim3(NU/16, 2), 128, 0, stream>>>(agg_u, WnuP, bnu, hu, 256, 128, 1);
    k_gemm_node<256><<<dim3(NV/16, 2), 128, 0, stream>>>(agg_v, WnvP, bnv, hv, 256, 128, 1);
}
